// Spatial_Attention_71803263254577
// MI455X (gfx1250) — matrix operand order repaired and checked by reading, not compiled, not run on hardware
//
#include <hip/hip_runtime.h>
#include <hip/hip_bf16.h>

// Spatial attention refinement on MI455X (gfx1250), wave32 + WMMA f16->f32.
// Shapes: B=32, C=2048, H=48, W=24 -> N=1152, MID=128.
// All WMMA operands are pre-packed f16 (fp32 accumulate). LDS tiles are stored
// transposed [n][k] with odd-dword row strides so every fragment element-pair
// is one aligned dword; staging is vectorized (b128/b64 global -> b16 LDS) and
// double-buffered (1 barrier per K-step, loads overlap WMMAs).

typedef __attribute__((ext_vector_type(16))) _Float16 v16h;
typedef __attribute__((ext_vector_type(8)))  float    v8f;

union FragH  { unsigned int u[8]; v16h v; };
union Pack8H { uint4 u; _Float16 h[8]; };
union Pack4H { uint2 u; _Float16 h[4]; };

constexpr int Bb = 32;
constexpr int Cc = 2048;
constexpr int Nn = 1152;    // H*W
constexpr int Mm = 128;     // C/16

// ---------------------------------------------------------------------------
// Kernel 0: fp32 -> f16 conversion (cam, w1, w2). float4 in, b64 out.
// ---------------------------------------------------------------------------
__global__ void __launch_bounds__(256)
cvt_kernel(const float4* __restrict__ src, uint2* __restrict__ dst, int count4) {
  int idx = blockIdx.x * 256 + threadIdx.x;
  int stride = gridDim.x * 256;
  for (int i = idx; i < count4; i += stride) {
    float4 v = src[i];
    Pack4H p;
    p.h[0] = (_Float16)v.x; p.h[1] = (_Float16)v.y;
    p.h[2] = (_Float16)v.z; p.h[3] = (_Float16)v.w;
    dst[i] = p.u;
  }
}

// ---------------------------------------------------------------------------
// Kernel 1: f1 = w1 @ x, f2 = w2 @ x  (per batch [128,2048]@[2048,1152]), f16 out.
// 8 waves; wave w owns m-tile w. x K-tile staged transposed+converted in LDS
// (double-buffered); both WMMAs share the B fragment; A loads are b128 pairs.
// ---------------------------------------------------------------------------
__global__ void __launch_bounds__(256)
proj_kernel(const float* __restrict__ x,
            const _Float16* __restrict__ w1h, const _Float16* __restrict__ w2h,
            _Float16* __restrict__ f1h, _Float16* __restrict__ f2h) {
  const int nt   = blockIdx.x;           // 0..71
  const int b    = blockIdx.y;           // 0..31
  const int tid  = threadIdx.x;
  const int lane = tid & 31;
  const int m0   = (tid >> 5) * 16;
  const int n0   = nt * 16;
  const int li   = lane & 15;
  const int half = lane >> 4;

  __shared__ _Float16 xs[2][16][42];     // [buf][n][k], 21-dword rows
  const float* xb = x + (size_t)b * Cc * Nn;

  // staging: thread -> (k row, 2 consecutive n) : one b64 load, two b16 stores
  const int sk = tid >> 3;               // 0..31
  const int sn = (tid & 7) * 2;          // 0,2,..,14
  auto stage = [&](int k0, int buf) {
    const float* src = &xb[(size_t)(k0 + sk) * Nn + n0 + sn];
    float2 v = *(const float2*)src;
    xs[buf][sn][sk]     = (_Float16)v.x;
    xs[buf][sn + 1][sk] = (_Float16)v.y;
  };

  const unsigned int* w1p = (const unsigned int*)w1h;
  const unsigned int* w2p = (const unsigned int*)w2h;

  v8f acc1 = {}, acc2 = {};
  stage(0, 0);
  __syncthreads();
  for (int kt = 0; kt < Cc / 32; ++kt) {
    const int k0 = kt * 32;
    if (kt + 1 < Cc / 32) stage(k0 + 32, (kt + 1) & 1);

    FragH a1, a2, bf;
    const unsigned int* xrow = (const unsigned int*)&xs[kt & 1][li][0];
#pragma unroll
    for (int i = 0; i < 8; ++i) {
      // A 16x32 f16 layout: half selects K {0-7,16-23} vs {8-15,24-31}
      int kb = ((i >= 4) ? 16 : 0) + (i & 3) * 2 + half * 8;
      size_t ha = ((size_t)(m0 + li) * Cc + k0 + kb) >> 1;   // packed pair
      a1.u[i] = w1p[ha];
      a2.u[i] = w2p[ha];
      // B 32x16 f16 layout: same K order as A (half selects K {0-7,16-23} vs {8-15,24-31})
      bf.u[i] = xrow[kb >> 1];
    }
    acc1 = __builtin_amdgcn_wmma_f32_16x16x32_f16(false, a1.v, false, bf.v,
                                                  (short)0, acc1, false, false);
    acc2 = __builtin_amdgcn_wmma_f32_16x16x32_f16(false, a2.v, false, bf.v,
                                                  (short)0, acc2, false, false);
    __syncthreads();
  }

#pragma unroll
  for (int r = 0; r < 8; ++r) {
    int m = half * 8 + r;
    size_t idx = ((size_t)b * Mm + m0 + m) * Nn + n0 + li;
    f1h[idx] = (_Float16)acc1[r];
    f2h[idx] = (_Float16)acc2[r];
  }
}

// ---------------------------------------------------------------------------
// Kernel 2: S[b,i,j] = sum_m f2[b,m,i] * f1[b,m,j]  (K=128), fp32 logits out.
// Block covers one i-tile x 128 j's. Both operand panels staged transposed in
// LDS (f2: [16i][128k], f1: [128j][128k]); every fragment read is one dword.
// ---------------------------------------------------------------------------
__global__ void __launch_bounds__(256)
scores_kernel(const _Float16* __restrict__ f1h, const _Float16* __restrict__ f2h,
              float* __restrict__ S) {
  const int jg   = blockIdx.x;           // 0..8  (128-col j group)
  const int it   = blockIdx.y;           // 0..71
  const int b    = blockIdx.z;
  const int tid  = threadIdx.x;
  const int wave = tid >> 5;
  const int lane = tid & 31;
  const int i0   = it * 16;
  const int jb0  = jg * 128;
  const int li   = lane & 15;
  const int half = lane >> 4;

  __shared__ _Float16 f2s[16][134];      // [i][k], 67-dword rows
  __shared__ _Float16 f1s[128][134];     // [j][k]
  const _Float16* f1b = f1h + (size_t)b * Mm * Nn;
  const _Float16* f2b = f2h + (size_t)b * Mm * Nn;

  // stage f2 panel: 2048 halfs; thread -> one b128 (8 j/i at one k)
  {
    int k = tid >> 1, ib = (tid & 1) * 8;            // k 0..127
    Pack8H p;
    p.u = *(const uint4*)&f2b[(size_t)k * Nn + i0 + ib];
#pragma unroll
    for (int j = 0; j < 8; ++j) f2s[ib + j][k] = p.h[j];
  }
  // stage f1 panel: 16384 halfs; 8 b128 loads per thread
#pragma unroll
  for (int t = 0; t < 8; ++t) {
    int e = tid + t * 256;               // 0..2047
    int k = e >> 4, jb = (e & 15) * 8;
    Pack8H p;
    p.u = *(const uint4*)&f1b[(size_t)k * Nn + jb0 + jb];
#pragma unroll
    for (int j = 0; j < 8; ++j) f1s[jb + j][k] = p.h[j];
  }
  __syncthreads();

  v8f acc = {};
  const unsigned int* arow = (const unsigned int*)&f2s[li][0];
  const unsigned int* brow = (const unsigned int*)&f1s[wave * 16 + li][0];
#pragma unroll
  for (int k0 = 0; k0 < Mm; k0 += 32) {
    FragH a, bf;
#pragma unroll
    for (int i = 0; i < 8; ++i) {
      int kb = ((i >= 4) ? 16 : 0) + (i & 3) * 2 + half * 8;
      a.u[i]  = arow[(k0 + kb) >> 1];                // A = f2^T, packed pair
      bf.u[i] = brow[(k0 + kb) >> 1];
    }
    acc = __builtin_amdgcn_wmma_f32_16x16x32_f16(false, a.v, false, bf.v,
                                                 (short)0, acc, false, false);
  }

#pragma unroll
  for (int r = 0; r < 8; ++r) {
    int m = half * 8 + r;
    S[((size_t)b * Nn + i0 + m) * Nn + jb0 + wave * 16 + li] = acc[r];
  }
}

// ---------------------------------------------------------------------------
// Kernel 3: row softmax; reads fp32 logits, writes f16 probs IN PLACE (f16 row
// aliases its own fp32 row; exp values held in registers, barrier before write).
// ---------------------------------------------------------------------------
__global__ void __launch_bounds__(256)
softmax_kernel(float* __restrict__ S) {
  const int i = blockIdx.x;
  const int b = blockIdx.y;
  float* row = S + ((size_t)b * Nn + i) * Nn;
  _Float16* rowh = (_Float16*)row;       // block owns this row
  __shared__ float red[256];
  const int tid = threadIdx.x;

  float m = -INFINITY;
  for (int j = tid; j < Nn; j += 256) m = fmaxf(m, row[j]);
  red[tid] = m;
  __syncthreads();
  for (int s = 128; s > 0; s >>= 1) {
    if (tid < s) red[tid] = fmaxf(red[tid], red[tid + s]);
    __syncthreads();
  }
  const float rowmax = red[0];
  __syncthreads();

  float ev[5];
  float sum = 0.f;
  int c = 0;
  for (int j = tid; j < Nn; j += 256) {
    float e = __expf(row[j] - rowmax);
    ev[c++] = e;
    sum += e;
  }
  red[tid] = sum;
  __syncthreads();                        // all fp32 reads complete here
  for (int s = 128; s > 0; s >>= 1) {
    if (tid < s) red[tid] += red[tid + s];
    __syncthreads();
  }
  const float inv = 1.0f / red[0];
  c = 0;
  for (int j = tid; j < Nn; j += 256) rowh[j] = (_Float16)(ev[c++] * inv);
}

// ---------------------------------------------------------------------------
// Kernel 4 (dominant, 174 GF): out = alpha * (cam @ attn) + cam
// GEMM M=2048, N=1152, K=1152, all-f16 operands. Block tile 128x64: 8 waves,
// wave = 1 m-tile x 4 n-tiles (one A fragment feeds 4 WMMAs). attn K-tile
// staged transposed in double-buffered LDS; prefetch 2 K-steps ahead.
// ---------------------------------------------------------------------------
__global__ void __launch_bounds__(256)
refine_kernel(const _Float16* __restrict__ cam16,
              const float* __restrict__ cam,        // fp32 residual
              const float* __restrict__ S,          // f16 probs aliased inside
              const float* __restrict__ alpha_p, float* __restrict__ out) {
  const int ng   = blockIdx.x;           // 0..17  (64-col group)
  const int mg   = blockIdx.y;           // 0..15
  const int b    = blockIdx.z;
  const int tid  = threadIdx.x;
  const int lane = tid & 31;
  const int m0   = mg * 128 + (tid >> 5) * 16;
  const int n0   = ng * 64;
  const int li   = lane & 15;
  const int half = lane >> 4;

  __shared__ _Float16 bs[2][64][42];     // [buf][n][k], 21-dword rows
  const _Float16* attnh = (const _Float16*)S;        // row stride 2*Nn halfs
  const float*    camb  = cam + (size_t)b * Cc * Nn;
  const unsigned int* camp =
      (const unsigned int*)(cam16 + (size_t)b * Cc * Nn);

  // staging: thread -> (k row, 8 consecutive n) : one b128 load, 8 b16 stores
  const int sk = tid >> 3;               // 0..31
  const int sn = (tid & 7) * 8;          // 0,8,..,56
  const size_t srow = ((size_t)b * Nn + sk) * (size_t)(2 * Nn) + n0 + sn;
  auto stage = [&](int k0, int buf) {
    Pack8H p;
    p.u = *(const uint4*)&attnh[srow + (size_t)k0 * (2 * Nn)];
#pragma unroll
    for (int j = 0; j < 8; ++j) bs[buf][sn + j][sk] = p.h[j];
  };

  v8f acc0 = {}, acc1 = {}, acc2 = {}, acc3 = {};
  stage(0, 0);
  __syncthreads();
  for (int kt = 0; kt < Nn / 32; ++kt) {
    const int k0 = kt * 32;
    if (kt + 1 < Nn / 32) stage(k0 + 32, (kt + 1) & 1);
    if (kt + 2 < Nn / 32)
      __builtin_prefetch(&attnh[srow + (size_t)(k0 + 64) * (2 * Nn)], 0, 0);

    FragH a, b0, b1, b2, b3;
    const unsigned int* r0 = (const unsigned int*)&bs[kt & 1][li][0];
    const unsigned int* r1 = (const unsigned int*)&bs[kt & 1][16 + li][0];
    const unsigned int* r2 = (const unsigned int*)&bs[kt & 1][32 + li][0];
    const unsigned int* r3 = (const unsigned int*)&bs[kt & 1][48 + li][0];
#pragma unroll
    for (int i = 0; i < 8; ++i) {
      int kb = ((i >= 4) ? 16 : 0) + (i & 3) * 2 + half * 8;
      a.u[i] = camp[((size_t)(m0 + li) * Nn + k0 + kb) >> 1];
      int kk = kb >> 1;
      b0.u[i] = r0[kk];
      b1.u[i] = r1[kk];
      b2.u[i] = r2[kk];
      b3.u[i] = r3[kk];
    }
    acc0 = __builtin_amdgcn_wmma_f32_16x16x32_f16(false, a.v, false, b0.v,
                                                  (short)0, acc0, false, false);
    acc1 = __builtin_amdgcn_wmma_f32_16x16x32_f16(false, a.v, false, b1.v,
                                                  (short)0, acc1, false, false);
    acc2 = __builtin_amdgcn_wmma_f32_16x16x32_f16(false, a.v, false, b2.v,
                                                  (short)0, acc2, false, false);
    acc3 = __builtin_amdgcn_wmma_f32_16x16x32_f16(false, a.v, false, b3.v,
                                                  (short)0, acc3, false, false);
    __syncthreads();
  }

  const float alpha = alpha_p[0];
#pragma unroll
  for (int r = 0; r < 8; ++r) {
    int m = half * 8 + r;
    size_t rowoff = (size_t)(m0 + m) * Nn;
    size_t o = (size_t)b * Cc * Nn + rowoff;
    out[o + n0 +      li] = alpha * acc0[r] + camb[rowoff + n0 +      li];
    out[o + n0 + 16 + li] = alpha * acc1[r] + camb[rowoff + n0 + 16 + li];
    out[o + n0 + 32 + li] = alpha * acc2[r] + camb[rowoff + n0 + 32 + li];
    out[o + n0 + 48 + li] = alpha * acc3[r] + camb[rowoff + n0 + 48 + li];
  }
}

// ---------------------------------------------------------------------------
extern "C" void kernel_launch(void* const* d_in, const int* in_sizes, int n_in,
                              void* d_out, int out_size, void* d_ws, size_t ws_size,
                              hipStream_t stream) {
  const float* feat  = (const float*)d_in[0];   // [B,C,H,W]
  const float* cam   = (const float*)d_in[1];   // [B,C,H,W]
  const float* w1    = (const float*)d_in[2];   // [MID,C]
  const float* w2    = (const float*)d_in[3];   // [MID,C]
  const float* alpha = (const float*)d_in[4];   // scalar
  float* out = (float*)d_out;

  // Workspace carve (f16 copies + fp32 scores; probs alias scores in place).
  char* p = (char*)d_ws;
  _Float16* w1h = (_Float16*)p; p += (size_t)Mm * Cc * 2;        // 512 KB
  _Float16* w2h = (_Float16*)p; p += (size_t)Mm * Cc * 2;        // 512 KB
  _Float16* f1h = (_Float16*)p; p += (size_t)Bb * Mm * Nn * 2;   // 9.4 MB
  _Float16* f2h = (_Float16*)p; p += (size_t)Bb * Mm * Nn * 2;   // 9.4 MB
  _Float16* c16 = (_Float16*)p; p += (size_t)Bb * Cc * Nn * 2;   // 151 MB
  float*    S   = (float*)p;                                      // 170 MB

  const int camCnt4 = Bb * Cc * Nn / 4;  // 18,874,368 float4s
  const int wCnt4   = Mm * Cc / 4;       // 65,536 float4s

  cvt_kernel<<<dim3((camCnt4 + 2047) / 2048), 256, 0, stream>>>(
      (const float4*)cam, (uint2*)c16, camCnt4);
  cvt_kernel<<<dim3((wCnt4 + 2047) / 2048), 256, 0, stream>>>(
      (const float4*)w1, (uint2*)w1h, wCnt4);
  cvt_kernel<<<dim3((wCnt4 + 2047) / 2048), 256, 0, stream>>>(
      (const float4*)w2, (uint2*)w2h, wCnt4);

  proj_kernel   <<<dim3(Nn / 16, Bb),           256, 0, stream>>>(feat, w1h, w2h, f1h, f2h);
  scores_kernel <<<dim3(9, Nn / 16, Bb),        256, 0, stream>>>(f1h, f2h, S);
  softmax_kernel<<<dim3(Nn, Bb),                256, 0, stream>>>(S);
  refine_kernel <<<dim3(Nn / 64, Cc / 128, Bb), 256, 0, stream>>>(c16, cam, S, alpha, out);
}
